// HiLo1_44839458570962
// MI455X (gfx1250) — compile-verified, matrix operand order checked by reading, not run on hardware
//
#include <hip/hip_runtime.h>
#include <hip/hip_bf16.h>

// ---------------------------------------------------------------------------
// HiLo attention for MI455X (gfx1250): wave32 + v_wmma_f32_16x16x32_f16
// ---------------------------------------------------------------------------

typedef __attribute__((ext_vector_type(16))) _Float16 v16h;
typedef __attribute__((ext_vector_type(8)))  _Float16 v8h;
typedef __attribute__((ext_vector_type(4)))  _Float16 v4h;
typedef __attribute__((ext_vector_type(8)))  float    v8f;

#define HILO_H   96
#define HILO_W   96
#define HILO_N   (HILO_H * HILO_W)     // 9216
#define HILO_C   128
#define HILO_NS  (48 * 48)             // 2304 pooled tokens
#define HILO_HD  16
#define HILO_SCALE 0.25f               // 16^-0.5

__device__ __forceinline__ v16h cat8(v8h lo, v8h hi) {
    return __builtin_shufflevector(lo, hi, 0,1,2,3,4,5,6,7,8,9,10,11,12,13,14,15);
}

// ---------------------------------------------------------------------------
// float -> fp16 conversion
// ---------------------------------------------------------------------------
__global__ void k_f2h(const float* __restrict__ src, _Float16* __restrict__ dst, int n) {
    int i = blockIdx.x * blockDim.x + threadIdx.x;
    if (i < n) dst[i] = (_Float16)src[i];
}

// ---------------------------------------------------------------------------
// AvgPool2d(2,2): x (96,96,128) -> low (48,48,128) fp32 + fp16
// ---------------------------------------------------------------------------
__global__ void k_pool(const float* __restrict__ x,
                       float* __restrict__ lowf, _Float16* __restrict__ lowh) {
    int idx = blockIdx.x * blockDim.x + threadIdx.x;
    if (idx >= HILO_NS * HILO_C) return;
    int c  = idx & 127;
    int p  = idx >> 7;
    int ws = p % 48, hs = p / 48;
    const float* b = x + ((size_t)(2*hs) * HILO_W + 2*ws) * HILO_C + c;
    float s = 0.25f * (b[0] + b[HILO_C] + b[HILO_W*HILO_C] + b[HILO_W*HILO_C + HILO_C]);
    lowf[idx] = s;
    lowh[idx] = (_Float16)s;
}

// ---------------------------------------------------------------------------
// high = PixelShuffle(low) @ restore_w^T + restore_b - x, stored fp16 in
// window-major order: row = g*4 + t  (g = window, t = i*2+j token in window)
// ---------------------------------------------------------------------------
__global__ void k_high(const float* __restrict__ x, const float* __restrict__ lowf,
                       const float* __restrict__ rw, const float* __restrict__ rb,
                       _Float16* __restrict__ highh) {
    __shared__ float l2h[32];
    int n  = blockIdx.x;               // pixel 0..9215
    int hI = n / HILO_W, wI = n % HILO_W;
    int c  = threadIdx.x;              // 0..127
    int hs = hI >> 1, ws = wI >> 1;
    int t  = (hI & 1) * 2 + (wI & 1);
    if (c < 32)
        l2h[c] = lowf[((size_t)hs * 48 + ws) * HILO_C + c * 4 + t];
    __syncthreads();
    float acc = rb[c] - x[(size_t)n * HILO_C + c];
    const float* w = rw + c * 32;      // restore_w is (128, 32) row-major
    #pragma unroll
    for (int i = 0; i < 32; ++i) acc += l2h[i] * w[i];
    int g = hs * 48 + ws;
    highh[((size_t)g * 4 + t) * HILO_C + c] = (_Float16)acc;
}

// ---------------------------------------------------------------------------
// Generic WMMA GEMM: C(M,N) = A(M,K) @ W(N,K)^T  [+ bias, fp32 out]
// One wave per 16x16 tile. Fragment layout per CDNA5 ISA:
//   lane half h = lane>>4, elem j:  K = (j&7) + 8*(2*(j>>3) + h)
//   -> two contiguous 8-half (16B) loads per operand per k-step.
// ---------------------------------------------------------------------------
__global__ __launch_bounds__(32)
void k_gemm_wmma(const _Float16* __restrict__ A, const _Float16* __restrict__ W,
                 _Float16* __restrict__ Ch, float* __restrict__ Cf,
                 const float* __restrict__ bias,
                 int K, int lda, int ldw, int ldc, int coff) {
    const int m0   = blockIdx.x * 16;
    const int n0   = blockIdx.y * 16;
    const int lane = threadIdx.x;
    const int r    = lane & 15;
    const int hf   = lane >> 4;
    v8f acc = {};
    const _Float16* arow = A + (size_t)(m0 + r) * lda;
    const _Float16* wrow = W + (size_t)(n0 + r) * ldw;
    for (int kb = 0; kb < K; kb += 32) {
        v16h af = cat8(*(const v8h*)(arow + kb + 8*hf),
                       *(const v8h*)(arow + kb + 16 + 8*hf));
        v16h wf = cat8(*(const v8h*)(wrow + kb + 8*hf),
                       *(const v8h*)(wrow + kb + 16 + 8*hf));
        acc = __builtin_amdgcn_wmma_f32_16x16x32_f16(false, af, false, wf,
                                                     (short)0, acc, false, false);
    }
    // D element v -> row m0 + v + 8*hf, col n0 + r
    #pragma unroll
    for (int v = 0; v < 8; ++v) {
        int row = m0 + v + 8*hf;
        if (Cf) Cf[(size_t)row * ldc + coff + n0 + r] = acc[v] + bias[n0 + r];
        else    Ch[(size_t)row * ldc + n0 + r]        = (_Float16)acc[v];
    }
}

// ---------------------------------------------------------------------------
// Low-frequency flash attention: 9216 queries x 2304 keys, 4 heads, HD=16.
// Block = 4 waves (one WGP's worth); each wave owns one 16-query tile of the
// same head. Each 32-key K/V chunk is staged ONCE per block into LDS
// (V transposed during staging), then all 4 waves build their WMMA fragments
// from LDS: 4x less global K/V traffic than per-wave loads.
//   lq : (9216, 64)  col = head*16 + d
//   lkv: (2304, 128) k col = head*16 + d ; v col = 64 + head*16 + d
// ---------------------------------------------------------------------------
__global__ __launch_bounds__(128)
void k_low_attn(const _Float16* __restrict__ lq, const _Float16* __restrict__ lkv,
                _Float16* __restrict__ out) {
    __shared__ _Float16 Kl[32 * 16];        // Kl[kk*16 + d]   (row-major chunk)
    __shared__ _Float16 Vt[16 * 32];        // Vt[d*32 + kk]   (transposed chunk)
    __shared__ _Float16 Pl[4][16 * 32];     // per-wave softmax probabilities
    const int tid  = threadIdx.x;           // 0..127
    const int wav  = tid >> 5;              // 0..3
    const int lane = tid & 31;
    const int r    = lane & 15;
    const int hf   = lane >> 4;
    const int tile = blockIdx.x * 4 + wav;  // query tile 0..575
    const int head = blockIdx.y;            // 0..3
    const v8h z8 = {};

    const _Float16* qrow = lq + (size_t)(tile * 16 + r) * 64 + head * 16;
    v16h qf = cat8(*(const v8h*)(qrow + 8*hf), z8);   // d = 8*hf + j < 16

    v8f O = {};
    float m[8], l[8];
    #pragma unroll
    for (int v = 0; v < 8; ++v) { m[v] = -3.0e38f; l[v] = 0.f; }

    // staging role: 4 threads per key row, 4 halfs (8B) each
    const int skk = tid >> 2;               // key row within chunk 0..31
    const int sd  = (tid & 3) * 4;          // d offset 0,4,8,12

    for (int key0 = 0; key0 < HILO_NS; key0 += 32) {
        // ---- cooperative K/V staging (2KB) + prefetch of next chunk ----
        {
            const _Float16* src = lkv + (size_t)(key0 + skk) * HILO_C + head * 16;
            *(v4h*)(&Kl[skk * 16 + sd]) = *(const v4h*)(src + sd);
            v4h vv = *(const v4h*)(src + 64 + sd);
            #pragma unroll
            for (int e = 0; e < 4; ++e) Vt[(sd + e) * 32 + skk] = vv[e];
            if (key0 + 32 < HILO_NS)
                __builtin_prefetch(src + 32 * HILO_C, 0, 3);   // global_prefetch
        }
        __syncthreads();

        // ---- scores: S(16x32) = Q x K^T (two WMMAs, inner dim padded) ----
        v16h kf0 = cat8(*(const v8h*)(&Kl[r * 16 + 8*hf]), z8);
        v16h kf1 = cat8(*(const v8h*)(&Kl[(r + 16) * 16 + 8*hf]), z8);
        v8f s0 = {}, s1 = {};
        s0 = __builtin_amdgcn_wmma_f32_16x16x32_f16(false, qf, false, kf0,
                                                    (short)0, s0, false, false);
        s1 = __builtin_amdgcn_wmma_f32_16x16x32_f16(false, qf, false, kf1,
                                                    (short)0, s1, false, false);

        // ---- online softmax (rows live in 16-lane halves of the wave) ----
        #pragma unroll
        for (int v = 0; v < 8; ++v) {
            float a = s0[v] * HILO_SCALE, b = s1[v] * HILO_SCALE;
            float rm = fmaxf(a, b);
            for (int sh = 1; sh < 16; sh <<= 1) rm = fmaxf(rm, __shfl_xor(rm, sh, 16));
            float mn    = fmaxf(m[v], rm);
            float alpha = __expf(m[v] - mn);
            float p0 = __expf(a - mn);
            float p1 = __expf(b - mn);
            float rs = p0 + p1;
            for (int sh = 1; sh < 16; sh <<= 1) rs += __shfl_xor(rs, sh, 16);
            l[v] = l[v] * alpha + rs;
            m[v] = mn;
            O[v] *= alpha;
            int row = v + 8 * hf;
            Pl[wav][row * 32 + r]      = (_Float16)p0;   // per-wave buffer:
            Pl[wav][row * 32 + 16 + r] = (_Float16)p1;   // same-wave LDS RAW is in-order
        }

        // ---- O += P(16x32) x V(32x16) ----
        v16h pf = cat8(*(const v8h*)(&Pl[wav][r * 32 + 8*hf]),
                       *(const v8h*)(&Pl[wav][r * 32 + 16 + 8*hf]));
        v16h vf = cat8(*(const v8h*)(&Vt[r * 32 + 8*hf]),
                       *(const v8h*)(&Vt[r * 32 + 16 + 8*hf]));
        O = __builtin_amdgcn_wmma_f32_16x16x32_f16(false, pf, false, vf,
                                                   (short)0, O, false, false);
        __syncthreads();   // all waves done with Kl/Vt before restaging
    }

    #pragma unroll
    for (int v = 0; v < 8; ++v) {
        int row = tile * 16 + v + 8 * hf;
        out[(size_t)row * 64 + head * 16 + r] = (_Float16)(O[v] / l[v]);
    }
}

// ---------------------------------------------------------------------------
// High path: 2x2-window attention (4 tokens) — tiny, pure VALU.
// qkv: (2304*4, 192) window-major; q col hh*16+d, k col 64+..., v col 128+...
// Output scattered to pixel-major (9216, 64).
// ---------------------------------------------------------------------------
__global__ void k_high_attn(const _Float16* __restrict__ qkv,
                            _Float16* __restrict__ hout) {
    int idx = blockIdx.x * blockDim.x + threadIdx.x;
    if (idx >= HILO_NS * 16) return;
    int qt = idx & 3;
    int hh = (idx >> 2) & 3;
    int g  = idx >> 4;
    const _Float16* base = qkv + (size_t)g * 4 * 192;
    float q[16];
    #pragma unroll
    for (int d = 0; d < 16; ++d) q[d] = (float)base[qt * 192 + hh * 16 + d];
    float s[4];
    float mx = -3e38f;
    #pragma unroll
    for (int t = 0; t < 4; ++t) {
        float acc = 0.f;
        const _Float16* k = base + t * 192 + 64 + hh * 16;
        #pragma unroll
        for (int d = 0; d < 16; ++d) acc += q[d] * (float)k[d];
        s[t] = acc * HILO_SCALE;
        mx = fmaxf(mx, s[t]);
    }
    float den = 0.f;
    #pragma unroll
    for (int t = 0; t < 4; ++t) { s[t] = __expf(s[t] - mx); den += s[t]; }
    float inv = 1.f / den;
    int hg = g / 48, wg = g % 48;
    int n  = (hg * 2 + (qt >> 1)) * HILO_W + wg * 2 + (qt & 1);
    _Float16* o = hout + (size_t)n * 64 + hh * 16;
    #pragma unroll
    for (int d = 0; d < 16; ++d) {
        float acc = 0.f;
        #pragma unroll
        for (int t = 0; t < 4; ++t)
            acc += s[t] * (float)base[t * 192 + 128 + hh * 16 + d];
        o[d] = (_Float16)(acc * inv);
    }
}

// ---------------------------------------------------------------------------
// Host-side orchestration
// ---------------------------------------------------------------------------
static inline size_t alignup(size_t v) { return (v + 255) & ~size_t(255); }

extern "C" void kernel_launch(void* const* d_in, const int* in_sizes, int n_in,
                              void* d_out, int out_size, void* d_ws, size_t ws_size,
                              hipStream_t stream) {
    const float* x         = (const float*)d_in[0];
    const float* restore_w = (const float*)d_in[1];
    const float* restore_b = (const float*)d_in[2];
    const float* l_q_w     = (const float*)d_in[3];
    const float* l_kv_w    = (const float*)d_in[4];
    const float* l_proj_w  = (const float*)d_in[5];
    const float* l_proj_b  = (const float*)d_in[6];
    const float* h_qkv_w   = (const float*)d_in[7];
    const float* h_proj_w  = (const float*)d_in[8];
    const float* h_proj_b  = (const float*)d_in[9];
    float* out = (float*)d_out;

    // workspace carve-up
    char* p = (char*)d_ws;
    size_t off = 0;
    auto take = [&](size_t bytes) { char* q = p + off; off = alignup(off + bytes); return q; };
    float*    lowf   = (float*)   take((size_t)HILO_NS * HILO_C * 4);
    _Float16* lowh   = (_Float16*)take((size_t)HILO_NS * HILO_C * 2);
    _Float16* xh     = (_Float16*)take((size_t)HILO_N  * HILO_C * 2);
    _Float16* highh  = (_Float16*)take((size_t)HILO_N  * HILO_C * 2);
    _Float16* lqw    = (_Float16*)take((size_t)64  * 128 * 2);
    _Float16* lkvw   = (_Float16*)take((size_t)128 * 128 * 2);
    _Float16* hqkvw  = (_Float16*)take((size_t)192 * 128 * 2);
    _Float16* lprojw = (_Float16*)take((size_t)64  * 64  * 2);
    _Float16* hprojw = (_Float16*)take((size_t)64  * 64  * 2);
    _Float16* lqh    = (_Float16*)take((size_t)HILO_N  * 64  * 2);
    _Float16* lkvh   = (_Float16*)take((size_t)HILO_NS * 128 * 2);
    _Float16* hqkvh  = (_Float16*)take((size_t)HILO_N  * 192 * 2);
    _Float16* lattn  = (_Float16*)take((size_t)HILO_N  * 64  * 2);
    _Float16* houth  = (_Float16*)take((size_t)HILO_N  * 64  * 2);
    (void)ws_size; (void)in_sizes; (void)n_in; (void)out_size;

    // 1) pooled low-freq tokens
    k_pool<<<(HILO_NS * HILO_C + 255) / 256, 256, 0, stream>>>(x, lowf, lowh);

    // 2) high-freq features (window-major, fp16)
    k_high<<<HILO_N, 128, 0, stream>>>(x, lowf, restore_w, restore_b, highh);

    // 3) fp16 conversions
    k_f2h<<<(HILO_N * HILO_C + 255) / 256, 256, 0, stream>>>(x, xh, HILO_N * HILO_C);
    k_f2h<<<(64  * 128 + 255) / 256, 256, 0, stream>>>(l_q_w,    lqw,    64  * 128);
    k_f2h<<<(128 * 128 + 255) / 256, 256, 0, stream>>>(l_kv_w,   lkvw,   128 * 128);
    k_f2h<<<(192 * 128 + 255) / 256, 256, 0, stream>>>(h_qkv_w,  hqkvw,  192 * 128);
    k_f2h<<<(64  * 64  + 255) / 256, 256, 0, stream>>>(l_proj_w, lprojw, 64  * 64);
    k_f2h<<<(64  * 64  + 255) / 256, 256, 0, stream>>>(h_proj_w, hprojw, 64  * 64);

    // 4) input GEMMs (WMMA)
    k_gemm_wmma<<<dim3(HILO_N / 16, 64 / 16),  32, 0, stream>>>(
        xh,    lqw,   lqh,   nullptr, nullptr, 128, 128, 128, 64, 0);
    k_gemm_wmma<<<dim3(HILO_NS / 16, 128 / 16), 32, 0, stream>>>(
        lowh,  lkvw,  lkvh,  nullptr, nullptr, 128, 128, 128, 128, 0);
    k_gemm_wmma<<<dim3(HILO_N / 16, 192 / 16), 32, 0, stream>>>(
        highh, hqkvw, hqkvh, nullptr, nullptr, 128, 128, 128, 192, 0);

    // 5) attention
    k_low_attn<<<dim3(HILO_N / 64, 4), 128, 0, stream>>>(lqh, lkvh, lattn);
    k_high_attn<<<(HILO_NS * 16 + 255) / 256, 256, 0, stream>>>(hqkvh, houth);

    // 6) output projections straight into d_out (cols 0..63 low, 64..127 high)
    k_gemm_wmma<<<dim3(HILO_N / 16, 4), 32, 0, stream>>>(
        lattn, lprojw, nullptr, out, l_proj_b, 64, 64, 64, 128, 0);
    k_gemm_wmma<<<dim3(HILO_N / 16, 4), 32, 0, stream>>>(
        houth, hprojw, nullptr, out, h_proj_b, 64, 64, 64, 128, 64);
}
